// LSTM_26877905338550
// MI455X (gfx1250) — matrix operand order repaired and checked by reading, not compiled, not run on hardware
//
#include <hip/hip_runtime.h>
#include <hip/hip_bf16.h>

// ---------------------------------------------------------------------------
// Fused tiny-LSTM chain on CDNA5 matrix cores (gfx1250, wave32).
//
// Mapping: D = W x actT per layer. A-operand = weights (M = out features),
// B-operand = activations (N = 16 batch rows owned by the wave).
//  * Weights pre-swizzled once per block into A-fragment-major LDS:
//    each fragment = two aligned 16B loads per lane (conflict-free).
//  * Activations in a per-wave row-major LDS tile [16 rows x 200 halfs];
//    B fragments are two aligned 16B loads (B layout: lanes 0-15 K=0..15,
//    lanes 16-31 K=16..31).
//  * C layout (lane = batch row, VGPR = feature) -> every tile store is one
//    aligned ds_store_b128.
//  * Biases folded into an extra K column against a constant-1 feature.
//  * [hs,hs] concat folded into W0' = W0[:,:30]+W0[:,30:60].
//  * Gate weights permuted into 4 groups padded to 32 (2 M-tiles per group):
//    the four gates of a cell element land in the SAME lane's registers
//    (g[tc], g[2+tc], g[4+tc], g[6+tc]) -> cell update entirely in registers,
//    no gate LDS round-trip; padded features give sigmoid(0)*tanh(0)=0.
//  * tanh/sigmoid built from v_exp_f32 + v_rcp_f32 (no IEEE div sequences).
// Same-wave LDS ops are architecturally in-order -> no barriers in the chain.
// ---------------------------------------------------------------------------

typedef __attribute__((ext_vector_type(16))) _Float16 v16h;
typedef __attribute__((ext_vector_type(8)))  _Float16 v8h;
typedef __attribute__((ext_vector_type(4)))  _Float16 v4h;
typedef __attribute__((ext_vector_type(8)))  float    v8f;

#define NTHREADS 128
#define WAVES 4
#define ROWS_PER_BLOCK (WAVES * 16)
#define S_ACT 200        // halfs per activation-tile row (400B, 16B aligned)

// Activation-tile column regions (all multiples of 32)
#define COL_X 0          // x (0..7), 1.0 at col 8, zeros to 31  (static)
#define COL_H 32         // hidden 0..29, 1.0 at col 30, 0 at 31
#define COL_R0 64        // hs / a1 / a3
#define COL_R1 128       // a0 / a2

// Weight-fragment LDS offsets (halfs); each fragment = 512 halfs
#define OFF_WX 0         // ht_W[:, 0:8]  + ht_b @k=8   (2 frags)
#define OFF_WH 1024      // ht_W[:, 8:38]               (2 frags)
#define OFF_W0 2048      // m_W0[:,:30]+m_W0[:,30:60] + m_b0 @k=30 (2)
#define OFF_W1 3072      // m_W1 + m_b1 @k=30           (3)
#define OFF_W2 4608      // m_W2 + m_b2 @k=45           (6)
#define OFF_W3 7680      // m_W3 + m_b3 @k=40           (4)
#define OFF_WG 9728      // g_W permuted to 4x32 groups + g_b @k=30 (8)
#define OFF_P1 13824     // p1_W + p1_b @k=30           (1)
#define W_TOTAL 14336

__device__ __forceinline__ v16h cat8(v8h lo, v8h hi) {
    return __builtin_shufflevector(lo, hi, 0,1,2,3,4,5,6,7,8,9,10,11,12,13,14,15);
}
__device__ __forceinline__ v8f v8f_zero() {
    v8f c;
#pragma unroll
    for (int e = 0; e < 8; ++e) c[e] = 0.f;
    return c;
}

// ---- fast transcendentals: v_exp_f32 / v_rcp_f32, no IEEE-div sequences ----
__device__ __forceinline__ float fast_tanhf(float x) {
    // tanh(x) = 1 - 2/(exp(2x)+1);  exp(2x) = exp2(x * 2*log2(e))
    const float e = __builtin_amdgcn_exp2f(x * 2.885390081777927f);
    return fmaf(-2.f, __builtin_amdgcn_rcpf(e + 1.f), 1.f);
}
__device__ __forceinline__ float fast_sigf(float x) {
    const float e = __builtin_amdgcn_exp2f(x * -1.4426950408889634f);
    return __builtin_amdgcn_rcpf(e + 1.f);
}

// ---- weight staging: A-fragment-major, chunk-interleaved (16B lane stride) ----
template <typename F>
__device__ __forceinline__ void stage_frag(_Float16* dst, int Mtiles, int nK32,
                                           F getw, int tid) {
    const int total = Mtiles * nK32 * 512;
    for (int idx = tid; idx < total; idx += NTHREADS) {
        const int frag = idx >> 9;
        const int rem  = idx & 511;
        const int e    = ((rem >> 8) << 3) | (rem & 7);
        const int lane = (rem >> 3) & 31;
        const int mt   = frag / nK32, ks = frag % nK32;
        const int hi   = lane >> 4,  m  = lane & 15;
        const int of   = mt * 16 + m;
        const int k    = ks * 32 + hi * 16 + e;
        dst[idx] = (_Float16)getw(of, k);
    }
}

// ---- GEMM: accumulate C[mt] += W_frags x actT over nK32 ksteps ----
__device__ __forceinline__ void gemm_acc(const _Float16* sWf, int Mtiles, int nK32,
                                         const _Float16* act, int colB, int lane,
                                         v8f* C) {
    const int row = lane & 15, hi = lane >> 4;
    for (int ks = 0; ks < nK32; ++ks) {
        const _Float16* bp = act + row * S_ACT + colB + ks * 32 + hi * 16;
        const v16h b = cat8(*(const v8h*)bp, *(const v8h*)(bp + 8));
        for (int mt = 0; mt < Mtiles; ++mt) {
            const _Float16* ap = sWf + (mt * nK32 + ks) * 512 + lane * 8;
            const v16h a = cat8(*(const v8h*)ap, *(const v8h*)(ap + 256));
            C[mt] = __builtin_amdgcn_wmma_f32_16x16x32_f16(
                false, a, false, b, (short)0, C[mt], false, false);
        }
    }
}

// ---- store C tiles as f16 rows (optionally tanh; oneFeat column forced to 1.0) ----
__device__ __forceinline__ void store_tiles(_Float16* act, const v8f* C, int Mtiles,
                                            int colD, int oneFeat, bool doTanh,
                                            int lane) {
    const int row = lane & 15, hi = lane >> 4;
    for (int mt = 0; mt < Mtiles; ++mt) {
        v8h h;
#pragma unroll
        for (int e = 0; e < 8; ++e) {
            float v = C[mt][e];
            if (doTanh) v = fast_tanhf(v);
            const int feat = mt * 16 + hi * 8 + e;
            h[e] = (feat == oneFeat) ? (_Float16)1.0f : (_Float16)v;
        }
        *(v8h*)(act + row * S_ACT + colD + mt * 16 + hi * 8) = h;
    }
}

// zero a 16-row x 16-col region with one b128 store per lane
__device__ __forceinline__ void zero16(_Float16* act, int col, int lane) {
    const int row = lane & 15, hi = lane >> 4;
    v8h z;
#pragma unroll
    for (int e = 0; e < 8; ++e) z[e] = (_Float16)0.f;
    *(v8h*)(act + row * S_ACT + col + hi * 8) = z;
}

__global__ __launch_bounds__(NTHREADS)
void lstm_fused_kernel(const float* __restrict__ x,
                       const float* __restrict__ ht_W, const float* __restrict__ ht_b,
                       const float* __restrict__ m_W0, const float* __restrict__ m_b0,
                       const float* __restrict__ m_W1, const float* __restrict__ m_b1,
                       const float* __restrict__ m_W2, const float* __restrict__ m_b2,
                       const float* __restrict__ m_W3, const float* __restrict__ m_b3,
                       const float* __restrict__ g_W,  const float* __restrict__ g_b,
                       const float* __restrict__ p1_W, const float* __restrict__ p1_b,
                       const float* __restrict__ p2_W, const float* __restrict__ p2_b,
                       float* __restrict__ out, int Btot)
{
    __shared__ __align__(16) _Float16 sW[W_TOTAL];
    __shared__ __align__(16) _Float16 sAct[WAVES][16 * S_ACT];
    __shared__ float sP2[16];

    const int tid = threadIdx.x;
    // ---- stage weights (bias folded at k = realK against constant-1 feature) ----
    stage_frag(sW + OFF_WX, 2, 1, [&](int of, int k) {
        if (of >= 30) return 0.f;
        if (k < 8)  return ht_W[of * 38 + k];
        if (k == 8) return ht_b[of];
        return 0.f; }, tid);
    stage_frag(sW + OFF_WH, 2, 1, [&](int of, int k) {
        return (of < 30 && k < 30) ? ht_W[of * 38 + 8 + k] : 0.f; }, tid);
    stage_frag(sW + OFF_W0, 2, 1, [&](int of, int k) {
        if (of >= 30) return 0.f;
        if (k < 30)  return m_W0[of * 60 + k] + m_W0[of * 60 + 30 + k];
        if (k == 30) return m_b0[of];
        return 0.f; }, tid);
    stage_frag(sW + OFF_W1, 3, 1, [&](int of, int k) {
        if (of >= 45) return 0.f;
        if (k < 30)  return m_W1[of * 30 + k];
        if (k == 30) return m_b1[of];
        return 0.f; }, tid);
    stage_frag(sW + OFF_W2, 3, 2, [&](int of, int k) {
        if (of >= 40) return 0.f;
        if (k < 45)  return m_W2[of * 45 + k];
        if (k == 45) return m_b2[of];
        return 0.f; }, tid);
    stage_frag(sW + OFF_W3, 2, 2, [&](int of, int k) {
        if (of >= 30) return 0.f;
        if (k < 40)  return m_W3[of * 40 + k];
        if (k == 40) return m_b3[of];
        return 0.f; }, tid);
    stage_frag(sW + OFF_WG, 8, 1, [&](int of, int k) {
        const int grp = of >> 5, c = of & 31;       // 4 gate groups padded to 32
        if (c >= 30) return 0.f;
        if (k < 30)  return g_W[(grp * 30 + c) * 30 + k];
        if (k == 30) return g_b[grp * 30 + c];
        return 0.f; }, tid);
    stage_frag(sW + OFF_P1, 1, 1, [&](int of, int k) {
        if (of >= 10) return 0.f;
        if (k < 30)  return p1_W[of * 30 + k];
        if (k == 30) return p1_b[of];
        return 0.f; }, tid);
    if (tid < 16) sP2[tid] = (tid < 10) ? p2_W[tid] : 0.f;
    __syncthreads();

    const int wave = tid >> 5, lane = tid & 31;
    const int row = lane & 15, hi = lane >> 4;
    _Float16* act = sAct[wave];
    const long base = (long)blockIdx.x * ROWS_PER_BLOCK + wave * 16;

    // static X region: zeros in cols 0..31, constant-1 at col 8
    zero16(act, COL_X, lane);
    zero16(act, COL_X + 16, lane);
    if (lane < 16) act[lane * S_ACT + 8] = (_Float16)1.0f;

    // hidden/cell in C-fragment layout: hid[tc][e] = feature tc*16+hi*8+e, row = lane&15
    v8f hid[2], cel[2];
    hid[0] = v8f_zero(); hid[1] = v8f_zero();
    cel[0] = v8f_zero(); cel[1] = v8f_zero();

    for (int t = 0; t < 3; ++t) {
        // ---- stage x_t into cols 0..7 (b64 per lane, 2 lanes per row) ----
        {
            const int r = lane >> 1, c0 = (lane & 1) * 4;
            long rr = base + r; if (rr >= Btot) rr = Btot - 1;
            const float4 xv = *reinterpret_cast<const float4*>(x + rr * 24 + t * 8 + c0);
            v4h xh; xh[0] = (_Float16)xv.x; xh[1] = (_Float16)xv.y;
            xh[2] = (_Float16)xv.z; xh[3] = (_Float16)xv.w;
            *(v4h*)(act + r * S_ACT + c0) = xh;
        }
        // ---- hidden -> COL_H (1.0 at feature 30 feeds p1/bias columns) ----
        store_tiles(act, hid, 2, COL_H, 30, false, lane);

        // hs = Wx@x + Wh@h (+biases via 1-columns)
        v8f hs[2]; hs[0] = v8f_zero(); hs[1] = v8f_zero();
        gemm_acc(sW + OFF_WX, 2, 1, act, COL_X, lane, hs);
        gemm_acc(sW + OFF_WH, 2, 1, act, COL_H, lane, hs);
        store_tiles(act, hs, 2, COL_R0, 30, false, lane);

        // a0 = tanh(W0' @ hs)
        v8f a0[2]; a0[0] = v8f_zero(); a0[1] = v8f_zero();
        gemm_acc(sW + OFF_W0, 2, 1, act, COL_R0, lane, a0);
        store_tiles(act, a0, 2, COL_R1, 30, true, lane);

        // a1 = tanh(W1 @ a0)   (48 wide at COL_R0, pad 112..127 zeroed)
        v8f a1[3]; a1[0] = v8f_zero(); a1[1] = v8f_zero(); a1[2] = v8f_zero();
        gemm_acc(sW + OFF_W1, 3, 1, act, COL_R1, lane, a1);
        store_tiles(act, a1, 3, COL_R0, 45, true, lane);
        zero16(act, COL_R0 + 48, lane);

        // a2 = tanh(W2 @ a1)   (K=64 from COL_R0; out 48 wide at COL_R1, pad zeroed)
        v8f a2[3]; a2[0] = v8f_zero(); a2[1] = v8f_zero(); a2[2] = v8f_zero();
        gemm_acc(sW + OFF_W2, 3, 2, act, COL_R0, lane, a2);
        store_tiles(act, a2, 3, COL_R1, 40, true, lane);
        zero16(act, COL_R1 + 48, lane);

        // a3 = tanh(W3 @ a2)   (K=64 from COL_R1; out at COL_R0)
        v8f a3[2]; a3[0] = v8f_zero(); a3[1] = v8f_zero();
        gemm_acc(sW + OFF_W3, 2, 2, act, COL_R1, lane, a3);
        store_tiles(act, a3, 2, COL_R0, 30, true, lane);

        // gates z = G' @ a3 stay in REGISTERS: group grp, cell index c ->
        // fragment g[2*grp + tc][e] with c = tc*16 + hi*8 + e (same lane).
        v8f g[8];
#pragma unroll
        for (int i = 0; i < 8; ++i) g[i] = v8f_zero();
        gemm_acc(sW + OFF_WG, 8, 1, act, COL_R0, lane, g);

        // reference: gates = tanh(z); i,f,o = sigmoid(gate); g = tanh(gate)
#pragma unroll
        for (int tc = 0; tc < 2; ++tc) {
#pragma unroll
            for (int e = 0; e < 8; ++e) {
                const float ti = fast_tanhf(g[tc    ][e]);
                const float tf = fast_tanhf(g[2 + tc][e]);
                const float to = fast_tanhf(g[4 + tc][e]);
                const float tg = fast_tanhf(g[6 + tc][e]);
                const float cn = fast_sigf(tf) * cel[tc][e]
                               + fast_sigf(ti) * fast_tanhf(tg);
                cel[tc][e] = cn;
                hid[tc][e] = fast_sigf(to) * fast_tanhf(cn);
            }
        }
    }

    // ---- heads: p1 (30->10, bias via H's 1-column), p2 (10->1) ----
    store_tiles(act, hid, 2, COL_H, 30, false, lane);
    v8f o1[1]; o1[0] = v8f_zero();
    gemm_acc(sW + OFF_P1, 1, 1, act, COL_H, lane, o1);

    float part = 0.f;
#pragma unroll
    for (int e = 0; e < 8; ++e) part += o1[0][e] * sP2[hi * 8 + e];
    part += __shfl_xor(part, 16, 32);   // combine feature halves of the same row
    if (lane < 16) {
        const long r = base + lane;
        if (r < Btot) out[r] = part + p2_b[0];
    }
}

extern "C" void kernel_launch(void* const* d_in, const int* in_sizes, int n_in,
                              void* d_out, int out_size, void* d_ws, size_t ws_size,
                              hipStream_t stream) {
    (void)n_in; (void)d_ws; (void)ws_size; (void)out_size;
    const float* x    = (const float*)d_in[0];
    const float* ht_W = (const float*)d_in[1];
    const float* ht_b = (const float*)d_in[2];
    const float* m_W0 = (const float*)d_in[3];
    const float* m_b0 = (const float*)d_in[4];
    const float* m_W1 = (const float*)d_in[5];
    const float* m_b1 = (const float*)d_in[6];
    const float* m_W2 = (const float*)d_in[7];
    const float* m_b2 = (const float*)d_in[8];
    const float* m_W3 = (const float*)d_in[9];
    const float* m_b3 = (const float*)d_in[10];
    const float* g_W  = (const float*)d_in[11];
    const float* g_b  = (const float*)d_in[12];
    const float* p1_W = (const float*)d_in[13];
    const float* p1_b = (const float*)d_in[14];
    const float* p2_W = (const float*)d_in[15];
    const float* p2_b = (const float*)d_in[16];
    float* out = (float*)d_out;

    const int Btot   = in_sizes[0] / 24;  // x is [B, 3, 8]
    const int blocks = (Btot + ROWS_PER_BLOCK - 1) / ROWS_PER_BLOCK;

    lstm_fused_kernel<<<blocks, NTHREADS, 0, stream>>>(
        x, ht_W, ht_b, m_W0, m_b0, m_W1, m_b1, m_W2, m_b2, m_W3, m_b3,
        g_W, g_b, p1_W, p1_b, p2_W, p2_b, out, Btot);
}
